// LigerFusedLinearMSE_51848845197512
// MI455X (gfx1250) — matrix operand order repaired and checked by reading, not compiled, not run on hardware
//
#include <hip/hip_runtime.h>

#define N_DIM 4096
#define H_DIM 1024
#define V_DIM 16384

#define BM 128
#define BN 256
#define BK 32
#define LDK 40   // padded LDS row stride (elements) -> 80B rows, conflict-friendly

typedef __attribute__((ext_vector_type(16))) __bf16 v16bf;
typedef __attribute__((ext_vector_type(8)))  __bf16 v8bf;
typedef __attribute__((ext_vector_type(4)))  __bf16 v4bf;
typedef __attribute__((ext_vector_type(8)))  float  v8f;

static __device__ __forceinline__ v16bf combine16(v8bf a, v8bf b) {
  return __builtin_shufflevector(a, b, 0,1,2,3,4,5,6,7,8,9,10,11,12,13,14,15);
}

__global__ __launch_bounds__(256)
void flmse_main(const float* __restrict__ input,
                const float* __restrict__ weight,
                const float* __restrict__ target,
                const float* __restrict__ bias,
                float* __restrict__ partial) {
  __shared__ __bf16 Ah[BM * LDK];   // 10 KB
  __shared__ __bf16 Al[BM * LDK];   // 10 KB
  __shared__ __bf16 Bh[BN * LDK];   // 20 KB
  __shared__ __bf16 Bl[BN * LDK];   // 20 KB
  __shared__ float  red[256];

  const int tid  = threadIdx.x;
  const int lane = tid & 31;
  const int wave = tid >> 5;
  const int wm   = wave & 1;   // 2 waves along M (64 rows each)
  const int wn   = wave >> 1;  // 4 waves along N (64 cols each)

  const int blockN0 = blockIdx.x * BN;
  const int blockM0 = blockIdx.y * BM;

  // ---- global->LDS staging: A = 1024 float4 (4/thread), B = 2048 float4 (8/thread)
  const int rowbase = tid >> 3;        // 0..31; rows rowbase + 32*i
  const int colb    = (tid & 7) * 4;   // float4 column

  const float* Abase = input  + (size_t)blockM0 * H_DIM;
  const float* Bbase = weight + (size_t)blockN0 * H_DIM;

  float4 pa[4], pb[8];

  auto gload = [&](const float* __restrict__ base, int k0, float4* dst, int cnt) {
    for (int i = 0; i < cnt; ++i) {
      const int r = rowbase + 32 * i;
      dst[i] = *(const float4*)(base + (size_t)r * H_DIM + k0 + colb);
    }
  };

  auto lstore = [&](__bf16* __restrict__ hiT, __bf16* __restrict__ loT,
                    const float4* src, int cnt) {
    for (int i = 0; i < cnt; ++i) {
      const int r = rowbase + 32 * i;
      const float v[4] = { src[i].x, src[i].y, src[i].z, src[i].w };
      v4bf h4, l4;
      #pragma unroll
      for (int j = 0; j < 4; ++j) {
        __bf16 h = (__bf16)v[j];
        float  rem = v[j] - (float)h;  // exact split
        h4[j] = h;
        l4[j] = (__bf16)rem;
      }
      *(v4bf*)(hiT + r * LDK + colb) = h4;   // 8B ds_store
      *(v4bf*)(loT + r * LDK + colb) = l4;
    }
  };

  // ---- WMMA fragment addressing (per ISA VGPR layout tables)
  const int m16  = lane & 15;
  const int hi16 = lane >> 4;                         // 0 or 1
  const int aoff = (wm * 64 + m16) * LDK + hi16 * 8;  // A: kh=8*hi16, halves at +0,+16
  const int boff = (wn * 64 + m16) * LDK + hi16 * 8;  // B: kh=8*hi16, halves at +0,+16

  v8f acc[4][4];
  const v8f vzero = { 0.f, 0.f, 0.f, 0.f, 0.f, 0.f, 0.f, 0.f };
  #pragma unroll
  for (int mi = 0; mi < 4; ++mi)
    #pragma unroll
    for (int ni = 0; ni < 4; ++ni)
      acc[mi][ni] = vzero;

  gload(Abase, 0, pa, 4);
  gload(Bbase, 0, pb, 8);

  for (int kt = 0; kt < H_DIM / BK; ++kt) {
    __syncthreads();               // previous-iteration LDS reads done
    lstore(Ah, Al, pa, 4);
    lstore(Bh, Bl, pb, 8);
    __syncthreads();

    if (kt + 1 < H_DIM / BK) {     // prefetch next slab while WMMAs run
      gload(Abase, (kt + 1) * BK, pa, 4);
      gload(Bbase, (kt + 1) * BK, pb, 8);
    }

    v16bf ah[4], al[4], bh[4], bl[4];
    #pragma unroll
    for (int mi = 0; mi < 4; ++mi) {
      const __bf16* ph = Ah + aoff + mi * 16 * LDK;
      const __bf16* pl = Al + aoff + mi * 16 * LDK;
      ah[mi] = combine16(*(const v8bf*)ph, *(const v8bf*)(ph + 16));
      al[mi] = combine16(*(const v8bf*)pl, *(const v8bf*)(pl + 16));
    }
    #pragma unroll
    for (int ni = 0; ni < 4; ++ni) {
      const __bf16* ph = Bh + boff + ni * 16 * LDK;
      const __bf16* pl = Bl + boff + ni * 16 * LDK;
      bh[ni] = combine16(*(const v8bf*)ph, *(const v8bf*)(ph + 16));
      bl[ni] = combine16(*(const v8bf*)pl, *(const v8bf*)(pl + 16));
    }

    #pragma unroll
    for (int mi = 0; mi < 4; ++mi) {
      #pragma unroll
      for (int ni = 0; ni < 4; ++ni) {
        acc[mi][ni] = __builtin_amdgcn_wmma_f32_16x16x32_bf16(
            false, ah[mi], false, bh[ni], (short)0, acc[mi][ni], false, false);
        acc[mi][ni] = __builtin_amdgcn_wmma_f32_16x16x32_bf16(
            false, ah[mi], false, bl[ni], (short)0, acc[mi][ni], false, false);
        acc[mi][ni] = __builtin_amdgcn_wmma_f32_16x16x32_bf16(
            false, al[mi], false, bh[ni], (short)0, acc[mi][ni], false, false);
      }
    }
  }

  // ---- fused epilogue: (c + bias[n] - target[m,n])^2, local sum
  // C layout: reg r, lanes 0-15 -> M=r, N=lane; lanes 16-31 -> M=8+r, N=lane-16
  float lsum = 0.f;
  #pragma unroll
  for (int ni = 0; ni < 4; ++ni) {
    const int gn = blockN0 + wn * 64 + ni * 16 + (lane & 15);
    const float bv = bias[gn];
    #pragma unroll
    for (int mi = 0; mi < 4; ++mi) {
      const int gm0 = blockM0 + wm * 64 + mi * 16 + (hi16 << 3);
      #pragma unroll
      for (int r = 0; r < 8; ++r) {
        const float t = target[(size_t)(gm0 + r) * V_DIM + gn];
        const float d = acc[mi][ni][r] + bv - t;
        lsum += d * d;
      }
    }
  }

  // ---- workgroup reduce -> one partial per block
  red[tid] = lsum;
  __syncthreads();
  #pragma unroll
  for (int s = 128; s > 0; s >>= 1) {
    if (tid < s) red[tid] += red[tid + s];
    __syncthreads();
  }
  if (tid == 0) partial[blockIdx.y * gridDim.x + blockIdx.x] = red[0];
}

__global__ __launch_bounds__(256)
void flmse_reduce(const float* __restrict__ partial, float* __restrict__ out, int n) {
  __shared__ double red[256];
  double s = 0.0;
  for (int i = threadIdx.x; i < n; i += 256) s += (double)partial[i];
  red[threadIdx.x] = s;
  __syncthreads();
  for (int k = 128; k > 0; k >>= 1) {
    if ((int)threadIdx.x < k) red[threadIdx.x] += red[threadIdx.x + k];
    __syncthreads();
  }
  if (threadIdx.x == 0)
    out[0] = (float)(red[0] / ((double)N_DIM * (double)V_DIM));
}

extern "C" void kernel_launch(void* const* d_in, const int* in_sizes, int n_in,
                              void* d_out, int out_size, void* d_ws, size_t ws_size,
                              hipStream_t stream) {
  const float* input  = (const float*)d_in[0];
  const float* weight = (const float*)d_in[1];
  const float* target = (const float*)d_in[2];
  const float* bias   = (const float*)d_in[3];
  float* partial = (float*)d_ws;

  dim3 grid(V_DIM / BN, N_DIM / BM);   // (64, 32) = 2048 workgroups
  flmse_main<<<grid, 256, 0, stream>>>(input, weight, target, bias, partial);
  flmse_reduce<<<1, 256, 0, stream>>>(partial, (float*)d_out,
                                      (V_DIM / BN) * (N_DIM / BM));
}
